// TinyViTBlock_46926812676275
// MI455X (gfx1250) — matrix operand order repaired and checked by reading, not compiled, not run on hardware
//
#include <hip/hip_runtime.h>
#include <hip/hip_bf16.h>
#include <math.h>

// ---------------- problem constants ----------------
#define BB     64
#define HH     28
#define WW     28
#define CC     384
#define HEADS  12
#define WS     7
#define KD     32
#define NWIN   49          // 7*7 tokens per window
#define DHD    384         // KD*HEADS
#define MLPH   1536
#define LL     784         // H*W
#define NWINS  1024        // B * 4 * 4
#define MPAD   65536       // NWINS * 64 (windows padded to 64 rows)
#define MTOK   50176       // B * L
#define EPSL   1e-5f
#define VTP    80          // LDS pitch for v^T (multiple of 16 halves -> 32B rows)

typedef __attribute__((ext_vector_type(16))) _Float16 v16h;
typedef __attribute__((ext_vector_type(8)))  _Float16 v8h;
typedef __attribute__((ext_vector_type(8)))  float    v8f;

#define WMMA_F16(a, b, c) \
  __builtin_amdgcn_wmma_f32_16x16x32_f16(false, (a), false, (b), (short)0, (c), false, false)

// ---------------- WMMA fragment loaders (16x16x32 f16; contiguous k order, same in A and B) ----------------
// A matrix 16x32 f16: lane l holds row m = l%16; with g = l/16 the 16 halfs are
// K = {g*16 .. g*16+15}  -> two contiguous 16B chunks.
__device__ __forceinline__ v16h load_a_frag(const _Float16* A, int ld, int m0, int k0, int lane) {
  int m = m0 + (lane & 15);
  int g = lane >> 4;
  const _Float16* p = A + (size_t)m * ld + k0 + g * 16;
  v8h lo = *(const v8h*)p;
  v8h hi = *(const v8h*)(p + 8);
  v16h r;
#pragma unroll
  for (int i = 0; i < 8; ++i) { r[i] = lo[i]; r[8 + i] = hi[i]; }
  return r;
}

// B matrix 32x16 f16 with B[k][n] = W[n][k] (weights row-major (N,K)):
// lane l holds column n = n0 + l%16, K = (l/16)*16 + e  -> one contiguous 32B chunk.
__device__ __forceinline__ v16h load_b_frag(const _Float16* Wm, int ld, int n0, int k0, int lane) {
  int n  = n0 + (lane & 15);
  int kb = k0 + ((lane >> 4) << 4);
  return *(const v16h*)(Wm + (size_t)n * ld + kb);
}

// ---------------- weight fp32 -> f16 ----------------
__global__ void cvt_f32_f16(const float* __restrict__ s, _Float16* __restrict__ d, int n) {
  int i = blockIdx.x * 256 + threadIdx.x;
  if (i < n) d[i] = (_Float16)s[i];
}

// ---------------- expand relative-position bias to [12][64][64] with masks ----------------
// n >= 49 -> -1e30 (masked keys, exp -> 0); m >= 49 -> 0 (pad query rows, harmless).
__global__ void expand_bias(const float* __restrict__ attn_bias,
                            const int* __restrict__ bias_idxs,
                            float* __restrict__ bmat) {
  int h = blockIdx.x;
  for (int e = threadIdx.x; e < 64 * 64; e += 256) {
    int m = e >> 6, n = e & 63;
    float v;
    if (n >= NWIN)      v = -1e30f;
    else if (m < NWIN)  v = attn_bias[h * NWIN + bias_idxs[m * NWIN + n]];
    else                v = 0.0f;
    bmat[h * 4096 + e] = v;
  }
}

// ---------------- LayerNorm (optionally fused window partition) ----------------
template <bool WINDOWED>
__global__ __launch_bounds__(256) void ln_kernel(const float* __restrict__ x,
                                                 const float* __restrict__ g,
                                                 const float* __restrict__ b,
                                                 _Float16* __restrict__ out, int nrows) {
  int wave = threadIdx.x >> 5, lane = threadIdx.x & 31;
  int pr = blockIdx.x * 8 + wave;
  if (pr >= nrows) return;
  _Float16* dst = out + (size_t)pr * CC;
  const float* src;
  if (WINDOWED) {
    int row = pr & 63, win = pr >> 6;
    if (row >= NWIN) {  // zero pad rows (uniform per wave)
#pragma unroll
      for (int t = 0; t < 12; ++t) dst[t * 32 + lane] = (_Float16)0.0f;
      return;
    }
    int bz = win >> 4, wi = win & 15, wh = wi >> 2, wwc = wi & 3;
    int i = row / WS, j = row % WS;
    int l = (wh * WS + i) * WW + (wwc * WS + j);
    src = x + ((size_t)bz * LL + l) * CC;
  } else {
    src = x + (size_t)pr * CC;
  }
  float vals[12], s = 0.f;
#pragma unroll
  for (int t = 0; t < 12; ++t) { float v = src[t * 32 + lane]; vals[t] = v; s += v; }
#pragma unroll
  for (int off = 16; off > 0; off >>= 1) s += __shfl_xor(s, off);
  float mu = s * (1.f / CC);
  float q = 0.f;
#pragma unroll
  for (int t = 0; t < 12; ++t) { float d = vals[t] - mu; q += d * d; }
#pragma unroll
  for (int off = 16; off > 0; off >>= 1) q += __shfl_xor(q, off);
  float rstd = rsqrtf(q * (1.f / CC) + EPSL);
#pragma unroll
  for (int t = 0; t < 12; ++t) {
    int c = t * 32 + lane;
    dst[c] = (_Float16)((vals[t] - mu) * rstd * g[c] + b[c]);
  }
}

// ---------------- generic WMMA GEMM: out = A(MxK) * W(NxK)^T + bias ----------------
// Block = 8 waves; wave computes a 32x64 tile (2 A-frags, 4 B-frags, 8 WMMAs per
// K-step); block tile 256x64.
// EPI 0: f16 store (QKV).  EPI 1: exact GELU then f16 store (fc1).
// EPI 2: f32 out += acc (fc2 + residual already in out).
// EPI 3: un-window scatter, out = resid + bias + acc (proj), pad rows dropped.
template <int EPI>
__global__ __launch_bounds__(256) void gemm_wmma(const _Float16* __restrict__ A,
                                                 const _Float16* __restrict__ Wm,
                                                 const float* __restrict__ bias,
                                                 _Float16* __restrict__ outH,
                                                 float* __restrict__ outF,
                                                 const float* __restrict__ resid,
                                                 int M, int N, int K) {
  int lane = threadIdx.x & 31, wave = threadIdx.x >> 5;
  int m0 = blockIdx.y * 256 + wave * 32;
  int n0 = blockIdx.x * 64;
  v8f acc[2][4] = {};
  for (int k0 = 0; k0 < K; k0 += 32) {
    v16h a0 = load_a_frag(A, K, m0, k0, lane);
    v16h a1 = load_a_frag(A, K, m0 + 16, k0, lane);
#pragma unroll
    for (int nt = 0; nt < 4; ++nt) {
      v16h bfr = load_b_frag(Wm, K, n0 + nt * 16, k0, lane);
      acc[0][nt] = WMMA_F16(a0, bfr, acc[0][nt]);
      acc[1][nt] = WMMA_F16(a1, bfr, acc[1][nt]);
    }
  }
  int nl = lane & 15, mo = (lane >> 4) * 8;
#pragma unroll
  for (int mi = 0; mi < 2; ++mi) {
#pragma unroll
    for (int nt = 0; nt < 4; ++nt) {
#pragma unroll
      for (int r = 0; r < 8; ++r) {
        int m = m0 + mi * 16 + mo + r, n = n0 + nt * 16 + nl;
        float v = acc[mi][nt][r] + bias[n];
        if constexpr (EPI == 0) {
          outH[(size_t)m * N + n] = (_Float16)v;
        } else if constexpr (EPI == 1) {
          v = 0.5f * v * (1.0f + erff(v * 0.70710678118f));
          outH[(size_t)m * N + n] = (_Float16)v;
        } else if constexpr (EPI == 2) {
          size_t idx = (size_t)m * N + n;
          outF[idx] = outF[idx] + v;
        } else {
          int row = m & 63;
          if (row < NWIN) {
            int win = m >> 6, bz = win >> 4, wi = win & 15, wh = wi >> 2, wwc = wi & 3;
            int i = row / WS, j = row % WS;
            int l = (wh * WS + i) * WW + (wwc * WS + j);
            size_t idx = ((size_t)bz * LL + l) * CC + n;
            outF[idx] = resid[idx] + v;
          }
        }
      }
    }
  }
}

// ---------------- windowed attention: one block per (window, head) ----------------
// 4 waves, wave mt owns rows [mt*16, mt*16+16). v^T staged to LDS up front,
// S = q k^T (WMMA), pre-expanded bias add (branch-free), softmax with shfl_xor
// row-reductions, P staged in LDS as f16, then out = P v (WMMA from LDS).
__global__ __launch_bounds__(128) void attn_kernel(const _Float16* __restrict__ qkv,
                                                   const float* __restrict__ bmat,
                                                   _Float16* __restrict__ attn_out) {
  __shared__ _Float16 P[64 * 64];
  __shared__ _Float16 VT[KD * VTP];
  int win = blockIdx.x / HEADS, h = blockIdx.x % HEADS;
  int wave = threadIdx.x >> 5, lane = threadIdx.x & 31;
  const _Float16* base = qkv + (size_t)win * 64 * (3 * DHD);
  const int qoff = h * 96, koff = h * 96 + 32, voff = h * 96 + 64;
  const int mt = wave;
  const float scale = 0.17677669529663687f;  // 32^-0.5

  // stage v^T into LDS: VT[d][tok] (pitch 80 halves = 160B rows, 32B aligned)
  for (int e = threadIdx.x; e < 64 * KD; e += 128) {
    int tok = e >> 5, d = e & 31;
    VT[d * VTP + tok] = base[(size_t)tok * (3 * DHD) + voff + d];
  }

  // S = q @ k^T   (16x64 per wave, K = 32 in one WMMA step)
  v8f s[4] = {};
  v16h aq = load_a_frag(base + qoff, 3 * DHD, mt * 16, 0, lane);
#pragma unroll
  for (int nt = 0; nt < 4; ++nt) {
    v16h bk = load_b_frag(base + koff, 3 * DHD, nt * 16, 0, lane);
    s[nt] = WMMA_F16(aq, bk, s[nt]);
  }

  // branch-free bias add + softmax over the 64 padded key positions per row
  int nl = lane & 15, mb = mt * 16 + (lane >> 4) * 8;
  const float* bm = bmat + h * 4096;
#pragma unroll
  for (int r = 0; r < 8; ++r) {
    int m = mb + r;
    float mx = -1e30f;
#pragma unroll
    for (int nt = 0; nt < 4; ++nt) {
      float v = s[nt][r] * scale + bm[m * 64 + nt * 16 + nl];
      s[nt][r] = v;
      mx = fmaxf(mx, v);
    }
#pragma unroll
    for (int off = 8; off > 0; off >>= 1) mx = fmaxf(mx, __shfl_xor(mx, off));
    float sum = 0.f;
#pragma unroll
    for (int nt = 0; nt < 4; ++nt) { float e = __expf(s[nt][r] - mx); s[nt][r] = e; sum += e; }
#pragma unroll
    for (int off = 8; off > 0; off >>= 1) sum += __shfl_xor(sum, off);
    float inv = 1.f / sum;
#pragma unroll
    for (int nt = 0; nt < 4; ++nt)
      P[m * 64 + nt * 16 + nl] = (_Float16)(s[nt][r] * inv);
  }
  __syncthreads();  // covers VT staging; P is same-wave (DScnt-ordered)

  // out = P(64x64) @ v(64x32): two K-steps of 32, two 16-wide N tiles
  v8f o[2] = {};
#pragma unroll
  for (int ks = 0; ks < 2; ++ks) {
    v16h ap = load_a_frag(P, 64, mt * 16, ks * 32, lane);
#pragma unroll
    for (int nt = 0; nt < 2; ++nt) {
      v16h bv = load_b_frag(VT, VTP, nt * 16, ks * 32, lane);
      o[nt] = WMMA_F16(ap, bv, o[nt]);
    }
  }
#pragma unroll
  for (int nt = 0; nt < 2; ++nt) {
#pragma unroll
    for (int r = 0; r < 8; ++r) {
      int m = mb + r;
      attn_out[((size_t)win * 64 + m) * DHD + h * KD + nt * 16 + nl] = (_Float16)o[nt][r];
    }
  }
}

// ---------------- depthwise 3x3 conv (SAME) + BN, writes x2 into d_out ----------------
__global__ __launch_bounds__(128) void dwconv_bn(const float* __restrict__ x1,
                                                 const float* __restrict__ w,
                                                 const float* __restrict__ bng,
                                                 const float* __restrict__ bnb,
                                                 const float* __restrict__ bnm,
                                                 const float* __restrict__ bnv,
                                                 float* __restrict__ out) {
  int p = blockIdx.x;
  int bz = p / LL, l = p % LL, i = l / WW, j = l % WW;
#pragma unroll
  for (int cc = 0; cc < 3; ++cc) {
    int c = threadIdx.x + cc * 128;
    float acc = 0.f;
#pragma unroll
    for (int di = 0; di < 3; ++di) {
      int ii = i + di - 1;
      if (ii < 0 || ii >= HH) continue;
#pragma unroll
      for (int dj = 0; dj < 3; ++dj) {
        int jj = j + dj - 1;
        if (jj < 0 || jj >= WW) continue;
        acc += x1[((size_t)bz * LL + ii * WW + jj) * CC + c] * w[c * 9 + di * 3 + dj];
      }
    }
    float y = (acc - bnm[c]) * rsqrtf(bnv[c] + EPSL) * bng[c] + bnb[c];
    out[((size_t)bz * LL + l) * CC + c] = y;
  }
}

// ---------------- launch ----------------
extern "C" void kernel_launch(void* const* d_in, const int* in_sizes, int n_in,
                              void* d_out, int out_size, void* d_ws, size_t ws_size,
                              hipStream_t stream) {
  const float* x       = (const float*)d_in[0];
  const float* n1g     = (const float*)d_in[1];
  const float* n1b     = (const float*)d_in[2];
  const float* qkv_w   = (const float*)d_in[3];
  const float* qkv_b   = (const float*)d_in[4];
  const float* a_bias  = (const float*)d_in[5];
  const int*   b_idx   = (const int*)d_in[6];
  const float* proj_w  = (const float*)d_in[7];
  const float* proj_b  = (const float*)d_in[8];
  const float* conv_w  = (const float*)d_in[9];
  const float* bn_g    = (const float*)d_in[10];
  const float* bn_b    = (const float*)d_in[11];
  const float* bn_m    = (const float*)d_in[12];
  const float* bn_v    = (const float*)d_in[13];
  const float* n2g     = (const float*)d_in[14];
  const float* n2b     = (const float*)d_in[15];
  const float* fc1_w   = (const float*)d_in[16];
  const float* fc1_b   = (const float*)d_in[17];
  const float* fc2_w   = (const float*)d_in[18];
  const float* fc2_b   = (const float*)d_in[19];
  float* out = (float*)d_out;

  // workspace layout (slots reused across phases)
  char* ws = (char*)d_ws;
  size_t o = 0;
  auto alloc = [&](size_t bytes) { size_t cur = o; o += (bytes + 255) & ~(size_t)255; return cur; };
  size_t oWQ  = alloc((size_t)(3 * DHD) * CC * 2);       // qkv_w f16
  size_t oWP  = alloc((size_t)CC * DHD * 2);             // proj_w f16
  size_t oW1  = alloc((size_t)MLPH * CC * 2);            // fc1_w f16
  size_t oW2  = alloc((size_t)CC * MLPH * 2);            // fc2_w f16
  size_t oBM  = alloc((size_t)HEADS * 64 * 64 * 4);      // expanded bias matrix f32
  size_t oA   = alloc((size_t)MTOK * MLPH * 2);          // slotA: qkv acts / h1
  size_t oB   = alloc((size_t)MPAD * CC * 2);            // slotB: xw / x2n
  size_t oC   = alloc((size_t)MPAD * CC * 2);            // slotC: attn out
  size_t oD   = alloc((size_t)MTOK * CC * 4);            // slotD: x1 (f32)

  _Float16* wQ  = (_Float16*)(ws + oWQ);
  _Float16* wP  = (_Float16*)(ws + oWP);
  _Float16* w1  = (_Float16*)(ws + oW1);
  _Float16* w2  = (_Float16*)(ws + oW2);
  float*    bm  = (float*)(ws + oBM);
  _Float16* qkv = (_Float16*)(ws + oA);   // [MPAD][1152] f16
  _Float16* h1  = (_Float16*)(ws + oA);   // [MTOK][1536] f16 (reuse after attn)
  _Float16* xw  = (_Float16*)(ws + oB);   // [MPAD][384]  f16
  _Float16* x2n = (_Float16*)(ws + oB);   // [MTOK][384]  f16 (reuse)
  _Float16* att = (_Float16*)(ws + oC);   // [MPAD][384]  f16
  float*    x1  = (float*)(ws + oD);      // [MTOK][384]  f32

  // 0) weights -> f16, expand attention bias
  {
    int n;
    n = 3 * DHD * CC;  cvt_f32_f16<<<(n + 255) / 256, 256, 0, stream>>>(qkv_w, wQ, n);
    n = CC * DHD;      cvt_f32_f16<<<(n + 255) / 256, 256, 0, stream>>>(proj_w, wP, n);
    n = MLPH * CC;     cvt_f32_f16<<<(n + 255) / 256, 256, 0, stream>>>(fc1_w, w1, n);
    n = CC * MLPH;     cvt_f32_f16<<<(n + 255) / 256, 256, 0, stream>>>(fc2_w, w2, n);
    expand_bias<<<HEADS, 256, 0, stream>>>(a_bias, b_idx, bm);
  }
  // 1) LN1 + window partition (padded rows zeroed)
  ln_kernel<true><<<MPAD / 8, 256, 0, stream>>>(x, n1g, n1b, xw, MPAD);
  // 2) QKV GEMM: 65536 x 1152 x 384
  gemm_wmma<0><<<dim3((3 * DHD) / 64, MPAD / 256), 256, 0, stream>>>(
      xw, wQ, qkv_b, qkv, nullptr, nullptr, MPAD, 3 * DHD, CC);
  // 3) windowed attention
  attn_kernel<<<NWINS * HEADS, 128, 0, stream>>>(qkv, bm, att);
  // 4) proj GEMM + un-window + residual: 65536 x 384 x 384 -> x1
  gemm_wmma<3><<<dim3(CC / 64, MPAD / 256), 256, 0, stream>>>(
      att, wP, proj_b, nullptr, x1, x, MPAD, CC, DHD);
  // 5) depthwise conv + BN -> d_out (= x2)
  dwconv_bn<<<BB * LL, 128, 0, stream>>>(x1, conv_w, bn_g, bn_b, bn_m, bn_v, out);
  // 6) LN2 -> x2n (f16)
  ln_kernel<false><<<MTOK / 8, 256, 0, stream>>>(out, n2g, n2b, x2n, MTOK);
  // 7) fc1 + GELU: 50176 x 1536 x 384 -> h1
  gemm_wmma<1><<<dim3(MLPH / 64, MTOK / 256), 256, 0, stream>>>(
      x2n, w1, fc1_b, h1, nullptr, nullptr, MTOK, MLPH, CC);
  // 8) fc2 + bias + residual (d_out += ...): 50176 x 384 x 1536
  gemm_wmma<2><<<dim3(CC / 64, MTOK / 256), 256, 0, stream>>>(
      h1, w2, fc2_b, nullptr, out, nullptr, MTOK, CC, MLPH);
}
